// ForexLSTM_55336358642924
// MI455X (gfx1250) — matrix operand order repaired and checked by reading, not compiled, not run on hardware
//
#include <hip/hip_runtime.h>
#include <hip/hip_bf16.h>

typedef __bf16 bf16_t;
typedef __attribute__((ext_vector_type(16))) __bf16 v16bf;
typedef __attribute__((ext_vector_type(8)))  __bf16 v8bf;
typedef __attribute__((ext_vector_type(8)))  float  v8f;
typedef __attribute__((ext_vector_type(4)))  unsigned int u32x4;
typedef __attribute__((ext_vector_type(8)))  int i32x8;
typedef __attribute__((ext_vector_type(4)))  int i32x4;

#define TLEN 512
#define BATCH 1024
#define HD 128
#define WG_ROWS 32
#define HPITCH 136        // h staging row pitch (128 + 8)

#if __has_builtin(__builtin_amdgcn_tensor_load_to_lds) && __has_builtin(__builtin_amdgcn_s_wait_tensorcnt)
#define USE_TDM 1
#else
#define USE_TDM 0
#endif

__device__ __forceinline__ float sigmoidf_(float x) { return 1.0f / (1.0f + __expf(-x)); }
__device__ __forceinline__ float tanhf_(float x)    { return 2.0f / (1.0f + __expf(-2.0f * x)) - 1.0f; }

// Build a 16-element bf16 fragment from two 16-byte LDS chunks (lo at p, hi at p+gap elems)
__device__ __forceinline__ v16bf load_frag16(const bf16_t* p, int gap) {
  v8bf lo = *(const v8bf*)p;
  v8bf hi = *(const v8bf*)(p + gap);
  v16bf r;
#pragma unroll
  for (int i = 0; i < 8; ++i) { r[i] = lo[i]; r[i + 8] = hi[i]; }
  return r;
}

#if USE_TDM
// TDM: async DMA of a (32 rows x tile_d0 bf16) tile into LDS with hardware row
// padding so the LDS pitch matches the A-fragment staging layout.
__device__ __forceinline__ void tdm_issue_tile(unsigned lds_off, unsigned long long gaddr,
                                               int tile_d0, int pad_interval_code,
                                               int pad_amount_code, int stride_elems) {
  u32x4 g0;
  g0.x = 1u;                                                     // count=1 (valid), no gather
  g0.y = lds_off;                                                // lds_addr (bytes)
  g0.z = (unsigned)(gaddr & 0xFFFFFFFFull);                      // global_addr[31:0]
  g0.w = (unsigned)((gaddr >> 32) & 0x01FFFFFFull) | (2u << 30); // global_addr[56:32], type=2
  i32x8 g1;
  g1[0] = (1 << 16)                      // data_size = 2 bytes
        | (1 << 20)                      // pad_enable
        | (pad_interval_code << 22)      // DWORDs between pads
        | (pad_amount_code << 25);       // pad DWORDs
  g1[1] = (int)((unsigned)(tile_d0 & 0xFFFF) << 16);  // tensor_dim0 (low 16 @ [63:48])
  g1[2] = (int)(32u << 16);                           // tensor_dim1 = 32 (low 16 @ [95:80])
  g1[3] = (int)((unsigned)(tile_d0 & 0xFFFF) << 16);  // tile_dim0 (@ [127:112])
  g1[4] = 32;                                         // tile_dim1 = 32; tile_dim2 = 0
  g1[5] = stride_elems;                               // tensor_dim0_stride (low 32 @ [191:160])
  g1[6] = 0;
  g1[7] = 0;
  i32x4 z4 = {0, 0, 0, 0};
#if defined(__clang_major__) && (__clang_major__ >= 23)
  i32x8 z8 = {0, 0, 0, 0, 0, 0, 0, 0};
  __builtin_amdgcn_tensor_load_to_lds(g0, g1, z4, z4, z8, 0);
#else
  __builtin_amdgcn_tensor_load_to_lds(g0, g1, z4, z4, 0);
#endif
}
#endif

// ---------------------------------------------------------------------------
// Prep: pack concatenated bf16 weights + fused biases into workspace.
// wcat0[n][k]: k<14 = Wih0, 14..31 = 0 (pad), 32..159 = Whh0    (K=160)
// wcat1[n][k]: k<128 = Wih1, 128..255 = Whh1                    (K=256)
// ---------------------------------------------------------------------------
__global__ void prep_kernel(const float* __restrict__ Wih0, const float* __restrict__ Whh0,
                            const float* __restrict__ bih0, const float* __restrict__ bhh0,
                            const float* __restrict__ Wih1, const float* __restrict__ Whh1,
                            const float* __restrict__ bih1, const float* __restrict__ bhh1,
                            bf16_t* __restrict__ wcat0, bf16_t* __restrict__ wcat1,
                            float* __restrict__ bias0, float* __restrict__ bias1) {
  const int n = blockIdx.x;          // gate row 0..511
  const int tid = threadIdx.x;       // 256 threads
  for (int k = tid; k < 160; k += 256) {
    float v = (k < 14) ? Wih0[n * 14 + k] : ((k < 32) ? 0.0f : Whh0[n * 128 + (k - 32)]);
    wcat0[(size_t)n * 160 + k] = (bf16_t)v;
  }
  for (int k = tid; k < 256; k += 256) {
    float v = (k < 128) ? Wih1[n * 128 + k] : Whh1[n * 128 + (k - 128)];
    wcat1[(size_t)n * 256 + k] = (bf16_t)v;
  }
  if (tid == 0) {
    bias0[n] = bih0[n] + bhh0[n];
    bias1[n] = bih1[n] + bhh1[n];
  }
}

// ---------------------------------------------------------------------------
// x -> padded bf16 [B*T][16] (cols 14,15 = 0) so layer 0 can use the same
// TDM tile pipeline as layer 1.
// ---------------------------------------------------------------------------
__global__ __launch_bounds__(256) void xpad_kernel(const float* __restrict__ x,
                                                   bf16_t* __restrict__ xp) {
  const size_t i = (size_t)blockIdx.x * 256 + threadIdx.x;   // row over B*T
  const float* src = x + i * 14;
  bf16_t* dst = xp + i * 16;
  bf16_t tmp[16];
#pragma unroll
  for (int p = 0; p < 7; ++p) {
    float2 v = *(const float2*)(src + p * 2);
    tmp[p * 2]     = (bf16_t)v.x;
    tmp[p * 2 + 1] = (bf16_t)v.y;
  }
  tmp[14] = (bf16_t)0.0f;
  tmp[15] = (bf16_t)0.0f;
  *(v8bf*)dst       = *(v8bf*)tmp;
  *(v8bf*)(dst + 8) = *(v8bf*)(tmp + 8);
}

// ---------------------------------------------------------------------------
// Persistent LSTM layer. One WG owns 32 batch rows for all 512 timesteps.
// Both layers: TDM double-buffered x staging + double-buffered h staging,
// ONE barrier per timestep. XW = x width in K (32 for L0, 128 for L1);
// K = XW + 128. Wave mapping: r = wave&1 (row tile), hg = wave>>1.
// ---------------------------------------------------------------------------
template <int XW, bool IS_L0>
__global__ __launch_bounds__(256) void lstm_layer_kernel(
    const bf16_t* __restrict__ src,   // L0: xpad [B,T,16]; L1: h1 [B,T,128]
    const bf16_t* __restrict__ wg,    // [512,KTOT] bf16
    const float* __restrict__ biasg,  // [512]
    bf16_t* __restrict__ hout,        // L0: [B,T,128] bf16
    float* __restrict__ lastout)      // L1: [B,128] f32
{
  constexpr int KTOT = XW + 128;
  constexpr int KPAD = KTOT + 8;               // weight row pitch
  constexpr int KCH  = KTOT / 32;
  constexpr int KXCH = XW / 32;                // chunks sourced from xbuf
  constexpr int SRCW = IS_L0 ? 16 : 128;       // global row width (elems)
  constexpr int XPITCH = IS_L0 ? 40 : 136;     // x staging row pitch
  constexpr int TDM_PI = IS_L0 ? 2 : 5;        // pad interval code (8 / 64 DWORDs)
  constexpr int TDM_PA = IS_L0 ? 11 : 3;       // pad amount code (12 / 4 DWORDs)

  extern __shared__ __align__(16) char smem_raw[];
  bf16_t* wsm   = (bf16_t*)smem_raw;                 // [512][KPAD]
  bf16_t* xbuf0 = wsm + (size_t)512 * KPAD;          // [32][XPITCH]
  bf16_t* xbuf1 = xbuf0 + WG_ROWS * XPITCH;          // [32][XPITCH]
  bf16_t* hbuf0 = xbuf1 + WG_ROWS * XPITCH;          // [32][HPITCH]
  bf16_t* hbuf1 = hbuf0 + WG_ROWS * HPITCH;          // [32][HPITCH]

  const int tid  = threadIdx.x;
  const int lane = tid & 31;
  const int wave = tid >> 5;
  const int r    = wave & 1;     // row tile (16 rows)
  const int hg   = wave >> 1;    // h-col tile group (2 tiles of 16)
  const int nl   = lane & 15;
  const int mhi  = lane >> 4;    // 0/1 -> rows e / e+8 in C layout
  const int b0   = blockIdx.x * WG_ROWS;

  // Stage weights into LDS once (padded rows, b128 copies)
  for (int row = tid; row < 512; row += 256) {
    const uint4* s = (const uint4*)(wg + (size_t)row * KTOT);
    uint4* d = (uint4*)(wsm + row * KPAD);
#pragma unroll
    for (int c = 0; c < (KTOT * 2) / 16; ++c) d[c] = s[c];
  }
  // Zero h staging (h_{-1} = 0) and, for L0, x staging pad columns 16..31.
  for (int i = tid; i < 2 * WG_ROWS * HPITCH; i += 256) hbuf0[i] = (bf16_t)0.0f;
  if (IS_L0) {
    for (int i = tid; i < 2 * WG_ROWS * XPITCH; i += 256) xbuf0[i] = (bf16_t)0.0f;
  }

  float bi[2][4];
#pragma unroll
  for (int ht = 0; ht < 2; ++ht)
#pragma unroll
    for (int g = 0; g < 4; ++g)
      bi[ht][g] = biasg[g * 128 + (hg * 2 + ht) * 16 + nl];

  float cst[2][8], hval[2][8];
#pragma unroll
  for (int ht = 0; ht < 2; ++ht)
#pragma unroll
    for (int e = 0; e < 8; ++e) { cst[ht][e] = 0.0f; hval[ht][e] = 0.0f; }

#if USE_TDM
  // Prime the pipeline: async-load x tile for t=0 into xbuf0.
  if (wave == 0) {
    tdm_issue_tile((unsigned)(size_t)(void*)xbuf0,
                   (unsigned long long)(size_t)src + ((size_t)b0 * TLEN) * SRCW * 2ull,
                   SRCW, TDM_PI, TDM_PA, TLEN * SRCW);
  }
  __builtin_amdgcn_s_wait_tensorcnt(0);
#endif
  __syncthreads();

  for (int t = 0; t < TLEN; ++t) {
    bf16_t* xcur = (t & 1) ? xbuf1 : xbuf0;
    bf16_t* hcur = (t & 1) ? hbuf1 : hbuf0;
    bf16_t* hnxt = (t & 1) ? hbuf0 : hbuf1;

#if USE_TDM
    // Prefetch NEXT step's x tile; overlaps this step's GEMM + cell update.
    if (wave == 0 && t + 1 < TLEN) {
      bf16_t* xnext = ((t + 1) & 1) ? xbuf1 : xbuf0;
      tdm_issue_tile((unsigned)(size_t)(void*)xnext,
                     (unsigned long long)(size_t)src +
                         (((size_t)b0 * TLEN + (t + 1)) * SRCW) * 2ull,
                     SRCW, TDM_PI, TDM_PA, TLEN * SRCW);
    }
#else
    // Fallback: synchronous staged copy of this step's x tile.
    for (int idx = tid; idx < WG_ROWS * (SRCW / 8); idx += 256) {
      int row = idx / (SRCW / 8), ch = idx % (SRCW / 8);
      uint4 v = *(const uint4*)(src + ((size_t)(b0 + row) * TLEN + t) * SRCW + ch * 8);
      *(uint4*)(xcur + row * XPITCH + ch * 8) = v;
    }
    __syncthreads();
#endif

    // ---- stream previous step's h (read-only hcur) out to global (layer 0) ----
    if (IS_L0 && t > 0) {
      const int crow = tid >> 3, cchb = tid & 7;
#pragma unroll
      for (int u = 0; u < 2; ++u) {
        int ch = cchb + u * 8;
        uint4 v = *(const uint4*)(hcur + crow * HPITCH + ch * 8);
        *(uint4*)(hout + ((size_t)(b0 + crow) * TLEN + (t - 1)) * HD + ch * 8) = v;
      }
    }

    // ---- gates = A @ Wcat^T  (bf16 WMMA, f32 accum) ----
    v8f acc[2][4];
#pragma unroll
    for (int ht = 0; ht < 2; ++ht)
#pragma unroll
      for (int g = 0; g < 4; ++g)
#pragma unroll
        for (int e = 0; e < 8; ++e) acc[ht][g][e] = 0.0f;

#pragma unroll
    for (int kc = 0; kc < KCH; ++kc) {
      const bf16_t* ap = (kc < KXCH)
          ? xcur + (r * 16 + nl) * XPITCH + kc * 32 + mhi * 8
          : hcur + (r * 16 + nl) * HPITCH + (kc - KXCH) * 32 + mhi * 8;
      v16bf afrag = load_frag16(ap, 16);
#pragma unroll
      for (int ht = 0; ht < 2; ++ht) {
#pragma unroll
        for (int g = 0; g < 4; ++g) {
          const int ncol = g * 128 + (hg * 2 + ht) * 16 + nl;
          v16bf bfrag = load_frag16(wsm + ncol * KPAD + kc * 32 + mhi * 8, 16);
          acc[ht][g] = __builtin_amdgcn_wmma_f32_16x16x32_bf16(
              false, afrag, false, bfrag, (short)0, acc[ht][g], false, false);
        }
      }
    }

    // ---- LSTM cell update (per-thread registers) ----
#pragma unroll
    for (int ht = 0; ht < 2; ++ht) {
#pragma unroll
      for (int e = 0; e < 8; ++e) {
        float ig = sigmoidf_(acc[ht][0][e] + bi[ht][0]);
        float fg = sigmoidf_(acc[ht][1][e] + bi[ht][1]);
        float gg = tanhf_(acc[ht][2][e] + bi[ht][2]);
        float og = sigmoidf_(acc[ht][3][e] + bi[ht][3]);
        float c = fg * cst[ht][e] + ig * gg;
        cst[ht][e] = c;
        hval[ht][e] = og * tanhf_(c);
      }
    }

    // ---- write new h into the OTHER h buffer (no read/write conflict) ----
#pragma unroll
    for (int ht = 0; ht < 2; ++ht) {
#pragma unroll
      for (int e = 0; e < 8; ++e) {
        int row = r * 16 + e + 8 * mhi;
        int col = (hg * 2 + ht) * 16 + nl;
        hnxt[row * HPITCH + col] = (bf16_t)hval[ht][e];
      }
    }

#if USE_TDM
    __builtin_amdgcn_s_wait_tensorcnt(0);  // drain x_{t+1} prefetch
#endif
    __syncthreads();   // single barrier: publishes h_t and x_{t+1}
  }

  if (IS_L0) {
    // h_{T-1} lives in hbuf[TLEN & 1] = hbuf0 (published by final barrier).
    const int crow = tid >> 3, cchb = tid & 7;
#pragma unroll
    for (int u = 0; u < 2; ++u) {
      int ch = cchb + u * 8;
      uint4 v = *(const uint4*)(hbuf0 + crow * HPITCH + ch * 8);
      *(uint4*)(hout + ((size_t)(b0 + crow) * TLEN + (TLEN - 1)) * HD + ch * 8) = v;
    }
  } else {
#pragma unroll
    for (int ht = 0; ht < 2; ++ht)
#pragma unroll
      for (int e = 0; e < 8; ++e) {
        int row = r * 16 + e + 8 * mhi;
        int col = (hg * 2 + ht) * 16 + nl;
        lastout[(size_t)(b0 + row) * HD + col] = hval[ht][e];
      }
  }
}

// ---------------------------------------------------------------------------
// Head part 1: batch-norm statistics over the batch dimension -> scale/shift
// ---------------------------------------------------------------------------
__global__ void head_stats_kernel(const float* __restrict__ last,
                                  const float* __restrict__ gamma,
                                  const float* __restrict__ beta,
                                  float* __restrict__ ss) {
  const int f = threadIdx.x;   // 128 threads
  float s = 0.0f, q = 0.0f;
  for (int b = 0; b < BATCH; ++b) {
    float v = last[(size_t)b * HD + f];
    s += v; q += v * v;
  }
  float mu  = s * (1.0f / BATCH);
  float var = q * (1.0f / BATCH) - mu * mu;
  float rstd = rsqrtf(var + 1e-5f);
  float sc = gamma[f] * rstd;
  ss[f] = sc;
  ss[HD + f] = beta[f] - mu * sc;
}

// ---------------------------------------------------------------------------
// Head part 2: per-row MLP  out = relu(norm @ W1^T + b1) @ W2^T + b2
// ---------------------------------------------------------------------------
__global__ __launch_bounds__(256) void head_mlp_kernel(
    const float* __restrict__ last, const float* __restrict__ ss,
    const float* __restrict__ W1, const float* __restrict__ b1,
    const float* __restrict__ W2, const float* __restrict__ b2,
    float* __restrict__ out) {
  __shared__ float W1s[32 * HD];
  __shared__ float b1s[32], W2s[32], ssS[2 * HD];
  const int tid = threadIdx.x;
  for (int i = tid; i < 32 * HD; i += 256) W1s[i] = W1[i];
  if (tid < 32) { b1s[tid] = b1[tid]; W2s[tid] = W2[tid]; }
  if (tid < 2 * HD) ssS[tid] = ss[tid];
  __syncthreads();

  const int row = blockIdx.x * 256 + tid;
  const float* lr = last + (size_t)row * HD;
  float z[32];
#pragma unroll
  for (int j = 0; j < 32; ++j) z[j] = 0.0f;
  for (int f = 0; f < HD; ++f) {
    float n = lr[f] * ssS[f] + ssS[HD + f];
#pragma unroll
    for (int j = 0; j < 32; ++j) z[j] += n * W1s[j * HD + f];
  }
  float o = b2[0];
#pragma unroll
  for (int j = 0; j < 32; ++j) o += fmaxf(z[j] + b1s[j], 0.0f) * W2s[j];
  out[row] = o;
}

// ---------------------------------------------------------------------------
extern "C" void kernel_launch(void* const* d_in, const int* in_sizes, int n_in,
                              void* d_out, int out_size, void* d_ws, size_t ws_size,
                              hipStream_t stream) {
  const float* x    = (const float*)d_in[0];
  const float* Wih0 = (const float*)d_in[1];
  const float* Whh0 = (const float*)d_in[2];
  const float* bih0 = (const float*)d_in[3];
  const float* bhh0 = (const float*)d_in[4];
  const float* Wih1 = (const float*)d_in[5];
  const float* Whh1 = (const float*)d_in[6];
  const float* bih1 = (const float*)d_in[7];
  const float* bhh1 = (const float*)d_in[8];
  const float* gamma = (const float*)d_in[9];
  const float* beta  = (const float*)d_in[10];
  const float* W1 = (const float*)d_in[11];
  const float* b1 = (const float*)d_in[12];
  const float* W2 = (const float*)d_in[13];
  const float* b2 = (const float*)d_in[14];

  char* ws = (char*)d_ws;
  size_t off = 0;
  bf16_t* h1    = (bf16_t*)(ws + off); off += (size_t)BATCH * TLEN * HD * 2;  // 134 MB
  bf16_t* xpad  = (bf16_t*)(ws + off); off += (size_t)BATCH * TLEN * 16 * 2;  // 16 MB
  bf16_t* wcat0 = (bf16_t*)(ws + off); off += (size_t)512 * 160 * 2;
  bf16_t* wcat1 = (bf16_t*)(ws + off); off += (size_t)512 * 256 * 2;
  float*  bias0 = (float*)(ws + off);  off += 512 * 4;
  float*  bias1 = (float*)(ws + off);  off += 512 * 4;
  float*  last  = (float*)(ws + off);  off += (size_t)BATCH * HD * 4;
  float*  ss    = (float*)(ws + off);  off += 2 * HD * 4;

  prep_kernel<<<512, 256, 0, stream>>>(Wih0, Whh0, bih0, bhh0, Wih1, Whh1, bih1, bhh1,
                                       wcat0, wcat1, bias0, bias1);
  xpad_kernel<<<(BATCH * TLEN) / 256, 256, 0, stream>>>(x, xpad);

  const size_t lds0 =
      ((size_t)512 * (160 + 8) + 2 * WG_ROWS * 40 + 2 * WG_ROWS * HPITCH) * 2;   // ~195 KB
  lstm_layer_kernel<32, true><<<BATCH / WG_ROWS, 256, lds0, stream>>>(
      xpad, wcat0, bias0, h1, nullptr);

  const size_t lds1 =
      ((size_t)512 * (256 + 8) + 2 * WG_ROWS * 136 + 2 * WG_ROWS * HPITCH) * 2;  // ~305 KB
  lstm_layer_kernel<128, false><<<BATCH / WG_ROWS, 256, lds1, stream>>>(
      h1, wcat1, bias1, nullptr, last);

  head_stats_kernel<<<1, 128, 0, stream>>>(last, gamma, beta, ss);
  head_mlp_kernel<<<BATCH / 256, 256, 0, stream>>>(last, ss, W1, b1, W2, b2, (float*)d_out);
}
